// MoETransformer_56951266345563
// MI455X (gfx1250) — matrix operand order repaired and checked by reading, not compiled, not run on hardware
//
#include <hip/hip_runtime.h>
#include <hip/hip_bf16.h>

// ---------------------------------------------------------------------------
// Model constants (from the reference)
// ---------------------------------------------------------------------------
constexpr int V_SZ  = 50304;
constexpr int D_SZ  = 1024;
constexpr int L_N   = 2;
constexpr int H_N   = 16;
constexpr int HKV_N = 8;
constexpr int HD_N  = 64;
constexpr int E_N   = 64;
constexpr int I_N   = 512;
constexpr int KTOP  = 2;
constexpr int B_N   = 2;
constexpr int S_N   = 1024;
constexpr int T_N   = B_N * S_N;           // 2048
constexpr int CAP   = T_N * KTOP / E_N;    // 64

typedef __attribute__((ext_vector_type(16))) _Float16 v16h;
typedef __attribute__((ext_vector_type(8)))  float    v8f;
typedef unsigned int u32x4 __attribute__((ext_vector_type(4)));
typedef int          i32x8 __attribute__((ext_vector_type(8)));
typedef int          i32x4 __attribute__((ext_vector_type(4)));

#define DEV __device__ __forceinline__

#if defined(__has_builtin)
#if __has_builtin(__builtin_amdgcn_tensor_load_to_lds) && \
    __has_builtin(__builtin_amdgcn_s_wait_tensorcnt)
#define USE_TDM 1
#endif
#endif
#ifndef USE_TDM
#define USE_TDM 0
#endif

// CDNA5 WMMA f16 layouts (ISA 7.12.2):
// A 16x32 f16: lane = M (lo/hi 16 lanes same M), interleaved K per half.
DEV int ka_of(int e, int hf) { return ((e & 8) << 1) + (hf << 3) + (e & 7); }
// B 32x16 f16: lane = N, same interleaved K per half as A.
DEV int kb_of(int e, int hf) { return ((e & 8) << 1) + (hf << 3) + (e & 7); }

DEV v8f wmma32(v16h a, v16h b, v8f c) {
  return __builtin_amdgcn_wmma_f32_16x16x32_f16(false, a, false, b, (short)0, c,
                                                false, false);
}

// Issue a TDM load of a 64x32 fp32 tile (row stride Kd elements) into LDS
// offset 0, with 1-DWORD padding after every 32 DWORDs (LDS row = 33 floats).
DEV void tdm_load_a_tile(const float* gsrc, int Kd) {
#if USE_TDM
  unsigned long long ga = (unsigned long long)gsrc;
  u32x4 g0;
  g0[0] = 1u;                                   // count=1, user mode
  g0[1] = 0u;                                   // lds_addr = 0 (tile at LDS base)
  g0[2] = (unsigned)ga;                         // global_addr[31:0]
  g0[3] = (unsigned)((ga >> 32) & 0x1FFFFFFu)   // global_addr[56:32]
          | (2u << 30);                         // type = 2 ("image")
  i32x8 g1;
  g1[0] = (int)((2u << 16)                      // data_size = 4 bytes
                | (1u << 20)                    // pad_enable
                | (4u << 22));                  // pad_interval = 32 DWORDs
                                                // pad_amount = 0 -> 1 DWORD
  g1[1] = (int)((32u & 0xFFFFu) << 16);         // tensor_dim0 = 32 (bits 79:48 lo)
  g1[2] = (int)((64u & 0xFFFFu) << 16);         // dim0 hi=0 | tensor_dim1 = 64
  g1[3] = (int)(32u << 16);                     // dim1 hi=0 | tile_dim0 = 32
  g1[4] = (int)64u;                             // tile_dim1 = 64, tile_dim2 = 0
  g1[5] = (int)(unsigned)Kd;                    // tensor_dim0_stride[31:0]
  g1[6] = 0;                                    // stride hi | dim1_stride lo
  g1[7] = 0;
  i32x4 gz = {0, 0, 0, 0};
#if __clang_major__ >= 23
  i32x8 g4 = {0, 0, 0, 0, 0, 0, 0, 0};
  __builtin_amdgcn_tensor_load_to_lds(g0, g1, gz, gz, g4, 0);
#else
  __builtin_amdgcn_tensor_load_to_lds(g0, g1, gz, gz, 0);
#endif
#else
  (void)gsrc; (void)Kd;
#endif
}

// ---------------------------------------------------------------------------
// Generic WMMA GEMM: C[M,N] = A[M,K] * B[K,N]  (B optionally transposed:
// B stored as [N,K]).  fp32 in / fp32 out, f16 multiply, f32 accumulate.
// Block tile 64x128, BK=32, 256 threads = 8 waves (2 wave-rows x 4 wave-cols).
// Each wave owns a 32x32 register tile = 4 WMMAs per k-step.
// A tile staged by the Tensor Data Mover (wave 0 issues, s_wait_tensorcnt),
// B tile staged by vector loads with on-the-fly f32->f16 conversion.
// Batched via blockIdx.z with element strides.
// ---------------------------------------------------------------------------
template <bool BT>
__global__ __launch_bounds__(256) void gemm_wmma(
    const float* __restrict__ A, const float* __restrict__ Bm,
    float* __restrict__ C, int M, int N, int Kd,
    long long bsA, long long bsB, long long bsC) {
  // Single merged LDS block so the A-tile region sits at LDS offset 0 (the
  // TDM descriptor takes a raw LDS byte address).
  __shared__ __align__(16) unsigned char smem_raw[64 * 33 * 4 + 32 * 136 * 2];
  float*    Asf = (float*)smem_raw;                      // 64 rows x 33 floats
  _Float16* Bs  = (_Float16*)(smem_raw + 64 * 33 * 4);   // 32 rows x 136 halves

  A  += (long long)blockIdx.z * bsA;
  Bm += (long long)blockIdx.z * bsB;
  C  += (long long)blockIdx.z * bsC;

  const int tid  = threadIdx.x;
  const int row0 = blockIdx.y * 64;
  const int col0 = blockIdx.x * 128;
  const int lane = tid & 31;
  const int w    = tid >> 5;
  const int wr   = w >> 2;     // 0..1 : 32-row slab
  const int wc   = w & 3;      // 0..3 : 32-col slab
  const int mn   = lane & 15;
  const int hf   = lane >> 4;

  v8f acc00 = {0.f,0.f,0.f,0.f,0.f,0.f,0.f,0.f};
  v8f acc01 = acc00, acc10 = acc00, acc11 = acc00;

  for (int k0 = 0; k0 < Kd; k0 += 32) {
#if USE_TDM
    // ---- wave 0: DMA the 64x32 fp32 A tile into LDS via the TDM ----
    if (tid < 32) tdm_load_a_tile(A + (long long)row0 * Kd + k0, Kd);
#else
    // ---- fallback: stage A tile with vector loads ----
    for (int i = tid; i < 64 * 32; i += 256) {
      int r = i >> 5, c = i & 31;
      int gr = row0 + r;
      Asf[r * 33 + c] = (gr < M) ? A[(long long)gr * Kd + (k0 + c)] : 0.f;
    }
#endif
    // ---- stage B tile 32x128 (16 elements / thread) ----
    if (BT) {
      // B stored [N,K]: walk K with consecutive lanes for coalescing
      for (int i = tid; i < 32 * 128; i += 256) {
        int r = i & 31, c = i >> 5;
        int gc = col0 + c;
        float v = (gc < N) ? Bm[(long long)gc * Kd + (k0 + r)] : 0.f;
        Bs[r * 136 + c] = (_Float16)v;
      }
    } else {
      // B stored [K,N]: walk N with consecutive lanes
      for (int i = tid; i < 32 * 128; i += 256) {
        int r = i >> 7, c = i & 127;
        int gc = col0 + c;
        float v = (gc < N) ? Bm[(long long)(k0 + r) * N + gc] : 0.f;
        Bs[r * 136 + c] = (_Float16)v;
      }
    }
    // speculative prefetch of next A k-tile (global_prefetch_b8)
    if (k0 + 32 < Kd) {
      int pr = row0 + (tid >> 2);
      if (pr < M) __builtin_prefetch(A + (long long)pr * Kd + (k0 + 32), 0, 1);
    }
#if USE_TDM
    if (tid < 32) __builtin_amdgcn_s_wait_tensorcnt(0);
#endif
    __syncthreads();

    v16h a0, a1, b0, b1;
#pragma unroll
    for (int e = 0; e < 16; ++e) {
      int ka = ka_of(e, hf);
      int kb = kb_of(e, hf);
      a0[e] = (_Float16)Asf[(wr * 32 + mn) * 33 + ka];
      a1[e] = (_Float16)Asf[(wr * 32 + 16 + mn) * 33 + ka];
      b0[e] = Bs[kb * 136 + wc * 32 + mn];
      b1[e] = Bs[kb * 136 + wc * 32 + 16 + mn];
    }
    acc00 = wmma32(a0, b0, acc00);
    acc01 = wmma32(a0, b1, acc01);
    acc10 = wmma32(a1, b0, acc10);
    acc11 = wmma32(a1, b1, acc11);
    __syncthreads();
  }

  // ---- store 2x2 register tile ----
#pragma unroll
  for (int r = 0; r < 8; ++r) {
    int gr0 = row0 + wr * 32 + hf * 8 + r;
    int gr1 = gr0 + 16;
    int gc0 = col0 + wc * 32 + mn;
    int gc1 = gc0 + 16;
    if (gr0 < M) {
      if (gc0 < N) C[(long long)gr0 * N + gc0] = acc00[r];
      if (gc1 < N) C[(long long)gr0 * N + gc1] = acc01[r];
    }
    if (gr1 < M) {
      if (gc0 < N) C[(long long)gr1 * N + gc0] = acc10[r];
      if (gc1 < N) C[(long long)gr1 * N + gc1] = acc11[r];
    }
  }
}

// ---------------------------------------------------------------------------
// Streaming causal GQA attention, one wave per (q-tile of 16, head, batch).
// Online softmax, QK^T and P*V through WMMA.
// q: [T, H*HD], k/v: [T, HKV*HD], o: [T, H*HD]
// ---------------------------------------------------------------------------
__global__ __launch_bounds__(32) void attn_kernel(
    const float* __restrict__ q, const float* __restrict__ kbuf,
    const float* __restrict__ vbuf, float* __restrict__ obuf) {
  __shared__ float    sS[16][32];
  __shared__ _Float16 sP[16][32];
  __shared__ float    sM[16], sL[16], sAl[16];

  const int q0   = blockIdx.x * 16;
  const int h    = blockIdx.y;
  const int b    = blockIdx.z;
  const int kvh  = h >> 1;  // H/HKV == 2
  const int lane = threadIdx.x;
  const int mn   = lane & 15;
  const int hf   = lane >> 4;
  const float scale = 0.125f;  // 1/sqrt(HD)

  // Q fragments (fold in softmax scale)
  const float* qrow = q + ((long long)(b * S_N + q0 + mn) * (H_N * HD_N)) + h * HD_N;
  v16h a_lo, a_hi;
#pragma unroll
  for (int e = 0; e < 16; ++e) {
    int kk = ka_of(e, hf);
    a_lo[e] = (_Float16)(qrow[kk] * scale);
    a_hi[e] = (_Float16)(qrow[kk + 32] * scale);
  }

  if (lane < 16) { sM[lane] = -3.0e38f; sL[lane] = 0.f; }
  v8f o0 = {0.f,0.f,0.f,0.f,0.f,0.f,0.f,0.f};
  v8f o1 = o0, o2 = o0, o3 = o0;
  __syncthreads();

  const int kend = q0 + 16;
  for (int t0 = 0; t0 < kend; t0 += 32) {
    // ---- scores for two groups of 16 keys ----
#pragma unroll
    for (int g = 0; g < 2; ++g) {
      int key = t0 + g * 16 + mn;
      const float* krow =
          kbuf + ((long long)(b * S_N + key) * (HKV_N * HD_N)) + kvh * HD_N;
      v16h b_lo, b_hi;
#pragma unroll
      for (int e = 0; e < 16; ++e) {
        int kk = kb_of(e, hf);
        b_lo[e] = (_Float16)krow[kk];
        b_hi[e] = (_Float16)krow[kk + 32];
      }
      v8f s = {0.f,0.f,0.f,0.f,0.f,0.f,0.f,0.f};
      s = wmma32(a_lo, b_lo, s);
      s = wmma32(a_hi, b_hi, s);
#pragma unroll
      for (int r = 0; r < 8; ++r) {
        int row  = hf * 8 + r;
        int kpos = t0 + g * 16 + mn;
        sS[row][g * 16 + mn] = (kpos <= q0 + row) ? s[r] : -1.0e30f;
      }
    }
    __syncthreads();

    // ---- row-wise online softmax (one lane per row) ----
    if (lane < 16) {
      float mold = sM[lane];
      float mx = mold;
      for (int c = 0; c < 32; ++c) mx = fmaxf(mx, sS[lane][c]);
      float alpha = expf(mold - mx);
      float sum = sL[lane] * alpha;
      for (int c = 0; c < 32; ++c) {
        float p = expf(sS[lane][c] - mx);
        sP[lane][c] = (_Float16)p;
        sum += p;
      }
      sM[lane] = mx; sL[lane] = sum; sAl[lane] = alpha;
    }
    __syncthreads();

    // rescale running output
#pragma unroll
    for (int r = 0; r < 8; ++r) {
      float al = sAl[hf * 8 + r];
      o0[r] *= al; o1[r] *= al; o2[r] *= al; o3[r] *= al;
    }

    // P as A-fragment
    v16h pa;
#pragma unroll
    for (int e = 0; e < 16; ++e) pa[e] = sP[mn][ka_of(e, hf)];

    // ---- o += P * V ----
#pragma unroll
    for (int g = 0; g < 4; ++g) {
      v16h vb;
#pragma unroll
      for (int e = 0; e < 16; ++e) {
        int kk = kb_of(e, hf);
        const float* vrow =
            vbuf + ((long long)(b * S_N + t0 + kk) * (HKV_N * HD_N)) + kvh * HD_N;
        vb[e] = (_Float16)vrow[g * 16 + mn];
      }
      if      (g == 0) o0 = wmma32(pa, vb, o0);
      else if (g == 1) o1 = wmma32(pa, vb, o1);
      else if (g == 2) o2 = wmma32(pa, vb, o2);
      else             o3 = wmma32(pa, vb, o3);
    }
    __syncthreads();
  }

#pragma unroll
  for (int r = 0; r < 8; ++r) {
    int row = hf * 8 + r;
    float inv = 1.f / sL[row];
    float* orow =
        obuf + ((long long)(b * S_N + q0 + row) * (H_N * HD_N)) + h * HD_N;
    orow[0 * 16 + mn] = o0[r] * inv;
    orow[1 * 16 + mn] = o1[r] * inv;
    orow[2 * 16 + mn] = o2[r] * inv;
    orow[3 * 16 + mn] = o3[r] * inv;
  }
}

// ---------------------------------------------------------------------------
// Small kernels
// ---------------------------------------------------------------------------
__global__ void k_gather_embed(const int* __restrict__ ids,
                               const float* __restrict__ embed,
                               float* __restrict__ x) {
  int t = blockIdx.x;
  long long src = (long long)ids[t] * D_SZ;
  for (int i = threadIdx.x; i < D_SZ; i += blockDim.x)
    x[(long long)t * D_SZ + i] = embed[src + i];
}

__global__ __launch_bounds__(256) void k_rmsnorm(const float* __restrict__ x,
                                                 const float* __restrict__ w,
                                                 float* __restrict__ out,
                                                 float eps) {
  __shared__ float red[256];
  const int row = blockIdx.x, tid = threadIdx.x;
  const float* xr = x + (long long)row * D_SZ;
  float s = 0.f;
  for (int i = tid; i < D_SZ; i += 256) { float v = xr[i]; s += v * v; }
  red[tid] = s;
  __syncthreads();
  for (int st = 128; st > 0; st >>= 1) {
    if (tid < st) red[tid] += red[tid + st];
    __syncthreads();
  }
  const float r = rsqrtf(red[0] / (float)D_SZ + eps);
  float* orow = out + (long long)row * D_SZ;
  for (int i = tid; i < D_SZ; i += 256) orow[i] = xr[i] * r * w[i];
}

// Per-(token,head) RMS norm over HD=64 + RoPE.  32 lanes = 32 rotation pairs.
__global__ __launch_bounds__(32) void k_qknorm_rope(float* __restrict__ x,
                                                    const float* __restrict__ w,
                                                    int nheads) {
  int idx = blockIdx.x;
  int t = idx / nheads, head = idx % nheads;
  int s = t % S_N;  // sequence position
  int lane = threadIdx.x;
  float* row = x + ((long long)t * nheads + head) * HD_N;
  float a = row[lane];
  float b = row[lane + 32];
  float ss = a * a + b * b;
  for (int off = 16; off > 0; off >>= 1) ss += __shfl_xor(ss, off, 32);
  float r = rsqrtf(ss / (float)HD_N + 1e-5f);
  a = a * r * w[lane];
  b = b * r * w[lane + 32];
  float freq = expf(-logf(10000.0f) * (float)lane / 32.0f);
  float ang = (float)s * freq;
  float c = cosf(ang), sn = sinf(ang);
  row[lane]      = a * c - b * sn;
  row[lane + 32] = a * sn + b * c;
}

__global__ void k_gate_mul(float* __restrict__ o, const float* __restrict__ g,
                           long long n) {
  long long i = (long long)blockIdx.x * blockDim.x + threadIdx.x;
  if (i < n) o[i] *= 1.f / (1.f + expf(-g[i]));
}

__global__ void k_add(float* __restrict__ x, const float* __restrict__ y,
                      long long n) {
  long long i = (long long)blockIdx.x * blockDim.x + threadIdx.x;
  if (i < n) x[i] += y[i];
}

__global__ void k_silu_mul(float* __restrict__ g, const float* __restrict__ u,
                           long long n) {
  long long i = (long long)blockIdx.x * blockDim.x + threadIdx.x;
  if (i < n) {
    float v = g[i];
    g[i] = (v / (1.f + expf(-v))) * u[i];
  }
}

__global__ void k_zero(float* __restrict__ p, long long n) {
  long long i = (long long)blockIdx.x * blockDim.x + threadIdx.x;
  if (i < n) p[i] = 0.f;
}

// softmax over E=64 router logits + top-2 with renormalized weights
__global__ void k_router(const float* __restrict__ lg, float* __restrict__ wts,
                         int* __restrict__ idx) {
  int t = blockIdx.x * blockDim.x + threadIdx.x;
  if (t >= T_N) return;
  const float* r = lg + (long long)t * E_N;
  float mx = -3.0e38f;
  for (int e = 0; e < E_N; ++e) mx = fmaxf(mx, r[e]);
  float sum = 0.f;
  for (int e = 0; e < E_N; ++e) sum += expf(r[e] - mx);
  float b1 = -1.f, b2 = -1.f;
  int i1 = 0, i2 = 0;
  for (int e = 0; e < E_N; ++e) {
    float p = expf(r[e] - mx) / sum;
    if (p > b1)      { b2 = b1; i2 = i1; b1 = p; i1 = e; }
    else if (p > b2) { b2 = p; i2 = e; }
  }
  float tot = b1 + b2;
  wts[t * 2 + 0] = b1 / tot;
  wts[t * 2 + 1] = b2 / tot;
  idx[t * 2 + 0] = i1;
  idx[t * 2 + 1] = i2;
}

// capacity assignment, reference order: k-major, then token order
__global__ void k_slots(const int* __restrict__ idx, int* __restrict__ slot) {
  int e = threadIdx.x;  // 64 threads, one per expert
  int cnt = 0;
  for (int kk = 0; kk < KTOP; ++kk)
    for (int t = 0; t < T_N; ++t) {
      int a = t * KTOP + kk;
      if (idx[a] == e) {
        slot[a] = (cnt < CAP) ? cnt : -1;
        ++cnt;
      }
    }
}

__global__ void k_scatter(const float* __restrict__ h,
                          const int* __restrict__ idx,
                          const int* __restrict__ slot,
                          float* __restrict__ ein) {
  int tk = blockIdx.x;
  int sl = slot[tk];
  if (sl < 0) return;
  int t = tk / KTOP;
  int e = idx[tk];
  const float* src = h + (long long)t * D_SZ;
  float* dst = ein + ((long long)e * CAP + sl) * D_SZ;
  for (int i = threadIdx.x; i < D_SZ; i += blockDim.x) dst[i] = src[i];
}

__global__ void k_combine(float* __restrict__ x, const float* __restrict__ eo,
                          const int* __restrict__ idx,
                          const int* __restrict__ slot,
                          const float* __restrict__ wts) {
  int t = blockIdx.x;
  float* xr = x + (long long)t * D_SZ;
  for (int i = threadIdx.x; i < D_SZ; i += blockDim.x) {
    float acc = 0.f;
    for (int kk = 0; kk < KTOP; ++kk) {
      int a = t * KTOP + kk;
      int sl = slot[a];
      if (sl >= 0) {
        int e = idx[a];
        acc += wts[a] * eo[((long long)e * CAP + sl) * D_SZ + i];
      }
    }
    xr[i] += acc;
  }
}

// ---------------------------------------------------------------------------
// Launcher
// ---------------------------------------------------------------------------
extern "C" void kernel_launch(void* const* d_in, const int* in_sizes, int n_in,
                              void* d_out, int out_size, void* d_ws,
                              size_t ws_size, hipStream_t stream) {
  (void)in_sizes; (void)n_in; (void)out_size; (void)ws_size;

  const int*   ids    = (const int*)d_in[0];
  const float* embed  = (const float*)d_in[1];
  const float* attnNW = (const float*)d_in[2];
  const float* ffnNW  = (const float*)d_in[3];
  const float* Wq     = (const float*)d_in[4];
  const float* Wk     = (const float*)d_in[5];
  const float* Wv     = (const float*)d_in[6];
  const float* Wo     = (const float*)d_in[7];
  const float* Wg     = (const float*)d_in[8];
  const float* qNW    = (const float*)d_in[9];
  const float* kNW    = (const float*)d_in[10];
  const float* rW     = (const float*)d_in[11];
  const float* Weg    = (const float*)d_in[12];
  const float* Weu    = (const float*)d_in[13];
  const float* Wed    = (const float*)d_in[14];
  const float* lnf    = (const float*)d_in[15];

  char* p = (char*)d_ws;
  auto alloc = [&](size_t elems) {
    float* r = (float*)p;
    p += elems * sizeof(float);
    return r;
  };
  float* x    = alloc((size_t)T_N * D_SZ);
  float* hb   = alloc((size_t)T_N * D_SZ);
  float* qb   = alloc((size_t)T_N * H_N * HD_N);
  float* kb   = alloc((size_t)T_N * HKV_N * HD_N);
  float* vb   = alloc((size_t)T_N * HKV_N * HD_N);
  float* ob   = alloc((size_t)T_N * H_N * HD_N);
  float* gp   = alloc((size_t)T_N * H_N * HD_N);
  float* tmp  = alloc((size_t)T_N * D_SZ);
  float* rlog = alloc((size_t)T_N * E_N);
  float* ein  = alloc((size_t)E_N * CAP * D_SZ);
  float* gbuf = alloc((size_t)E_N * CAP * I_N);
  float* ubuf = alloc((size_t)E_N * CAP * I_N);
  float* eob  = alloc((size_t)E_N * CAP * D_SZ);
  float* topw = alloc((size_t)T_N * KTOP);
  int* tidx = (int*)p; p += (size_t)T_N * KTOP * sizeof(int);
  int* slot = (int*)p; p += (size_t)T_N * KTOP * sizeof(int);

  auto gemm = [&](const float* A, const float* Bm, float* C, int M, int N,
                  int Kd, bool bt, int batch, long long ba, long long bb,
                  long long bc) {
    dim3 grid((N + 127) / 128, (M + 63) / 64, batch);
    if (bt)
      gemm_wmma<true><<<grid, 256, 0, stream>>>(A, Bm, C, M, N, Kd, ba, bb, bc);
    else
      gemm_wmma<false><<<grid, 256, 0, stream>>>(A, Bm, C, M, N, Kd, ba, bb, bc);
  };
  auto ew = [&](long long n) { return dim3((unsigned)((n + 255) / 256)); };

  k_gather_embed<<<T_N, 256, 0, stream>>>(ids, embed, x);

  for (int l = 0; l < L_N; ++l) {
    // ---- attention block ----
    k_rmsnorm<<<T_N, 256, 0, stream>>>(x, attnNW + (size_t)l * D_SZ, hb, 1e-6f);
    gemm(hb, Wq + (size_t)l * D_SZ * H_N * HD_N, qb, T_N, H_N * HD_N, D_SZ,
         false, 1, 0, 0, 0);
    gemm(hb, Wk + (size_t)l * D_SZ * HKV_N * HD_N, kb, T_N, HKV_N * HD_N, D_SZ,
         false, 1, 0, 0, 0);
    gemm(hb, Wv + (size_t)l * D_SZ * HKV_N * HD_N, vb, T_N, HKV_N * HD_N, D_SZ,
         false, 1, 0, 0, 0);
    k_qknorm_rope<<<T_N * H_N, 32, 0, stream>>>(qb, qNW + (size_t)l * HD_N, H_N);
    k_qknorm_rope<<<T_N * HKV_N, 32, 0, stream>>>(kb, kNW + (size_t)l * HD_N,
                                                  HKV_N);
    attn_kernel<<<dim3(S_N / 16, H_N, B_N), 32, 0, stream>>>(qb, kb, vb, ob);
    gemm(hb, Wg + (size_t)l * D_SZ * H_N * HD_N, gp, T_N, H_N * HD_N, D_SZ,
         false, 1, 0, 0, 0);
    k_gate_mul<<<ew((long long)T_N * H_N * HD_N), 256, 0, stream>>>(
        ob, gp, (long long)T_N * H_N * HD_N);
    gemm(ob, Wo + (size_t)l * H_N * HD_N * D_SZ, tmp, T_N, D_SZ, H_N * HD_N,
         false, 1, 0, 0, 0);
    k_add<<<ew((long long)T_N * D_SZ), 256, 0, stream>>>(x, tmp,
                                                         (long long)T_N * D_SZ);

    // ---- MoE block ----
    k_rmsnorm<<<T_N, 256, 0, stream>>>(x, ffnNW + (size_t)l * D_SZ, hb, 1e-6f);
    gemm(hb, rW + (size_t)l * D_SZ * E_N, rlog, T_N, E_N, D_SZ, false, 1, 0, 0,
         0);
    k_router<<<(T_N + 255) / 256, 256, 0, stream>>>(rlog, topw, tidx);
    k_slots<<<1, E_N, 0, stream>>>(tidx, slot);
    k_zero<<<ew((long long)E_N * CAP * D_SZ), 256, 0, stream>>>(
        ein, (long long)E_N * CAP * D_SZ);
    k_scatter<<<T_N * KTOP, 256, 0, stream>>>(hb, tidx, slot, ein);
    gemm(ein, Weg + (size_t)l * E_N * D_SZ * I_N, gbuf, CAP, I_N, D_SZ, false,
         E_N, (long long)CAP * D_SZ, (long long)D_SZ * I_N,
         (long long)CAP * I_N);
    gemm(ein, Weu + (size_t)l * E_N * D_SZ * I_N, ubuf, CAP, I_N, D_SZ, false,
         E_N, (long long)CAP * D_SZ, (long long)D_SZ * I_N,
         (long long)CAP * I_N);
    k_silu_mul<<<ew((long long)E_N * CAP * I_N), 256, 0, stream>>>(
        gbuf, ubuf, (long long)E_N * CAP * I_N);
    gemm(gbuf, Wed + (size_t)l * E_N * I_N * D_SZ, eob, CAP, D_SZ, I_N, false,
         E_N, (long long)CAP * I_N, (long long)I_N * D_SZ,
         (long long)CAP * D_SZ);
    k_combine<<<T_N, 256, 0, stream>>>(x, eob, tidx, slot, topw);
  }

  // ---- final norm + logits (B transposed: embed is [V, D]) ----
  k_rmsnorm<<<T_N, 256, 0, stream>>>(x, lnf, hb, 1e-6f);
  gemm(hb, embed, (float*)d_out, T_N, V_SZ, D_SZ, true, 1, 0, 0, 0);
}
